// TDID_12541304504681
// MI455X (gfx1250) — matrix operand order repaired and checked by reading, not compiled, not run on hardware
//
#include <hip/hip_runtime.h>
#include <hip/hip_bf16.h>
#include <math.h>
#include <stdint.h>

// ---------------- problem constants ----------------
#define BB    8
#define TT    2
#define CC    512
#define HH    38
#define WW    50
#define HWN   1900           // H*W
#define CIN   1024           // T*C (conv input channels, also concat width)
#define COUT  512
#define NKS   32             // K-steps per tap (CIN / 32)
#define CLS_TOTAL (BB*18*HWN)

typedef __attribute__((ext_vector_type(8)))  __bf16 v8bf;
typedef __attribute__((ext_vector_type(16))) __bf16 v16bf;
typedef __attribute__((ext_vector_type(8)))  float  v8f;
typedef unsigned int v4u  __attribute__((ext_vector_type(4)));
typedef int          v4i_ __attribute__((ext_vector_type(4)));
typedef int          v8i_ __attribute__((ext_vector_type(8)));

__device__ __forceinline__ v16bf cat8(v8bf lo, v8bf hi) {
  return __builtin_shufflevector(lo, hi, 0,1,2,3,4,5,6,7,8,9,10,11,12,13,14,15);
}

// TDM descriptor + issue: 2D tile [tile1 rows x 32 cin] of bf16, row stride CIN
// elements, LDS padded +16B every 64B so rows land at stride 80B (= BSTR*2).
__device__ __forceinline__ void tdm_load_B(const __bf16* g, unsigned lds_addr) {
  const uint64_t ga = (uint64_t)(uintptr_t)g;
  const v4u g0 = { 1u,                                   // count=1 (user D#)
                   lds_addr,                             // lds_addr
                   (unsigned)ga,                         // global_addr[31:0]
                   ((unsigned)(ga >> 32) & 0x01ffffffu)  // global_addr[56:32]
                     | (2u << 30) };                     // type=2 (image)
  const v8i_ g1 = {
    (int)((1u << 16) |            // data_size = 1 -> 2 bytes
          (1u << 20) |            // pad_enable
          (3u << 22) |            // pad_interval: 16 DWORDs (64B)
          (3u << 25)),            // pad_amount:   4 DWORDs (16B)
    (int)(((unsigned)CIN & 0xffffu) << 16),                    // tensor_dim0 lo16
    (int)((((unsigned)CIN >> 16) & 0xffffu) | (4608u << 16)),  // dim0 hi16 | dim1 lo16
    (int)(32u << 16),                                          // dim1 hi16=0 | tile_dim0=32
    (int)128u,                                                 // tile_dim1=128, tile_dim2=0
    (int)CIN,                                                  // tensor_dim0_stride lo32
    0, 0 };
  const v4i_ gz = { 0, 0, 0, 0 };
#if __clang_major__ >= 23
  const v8i_ gz8 = { 0, 0, 0, 0, 0, 0, 0, 0 };
  __builtin_amdgcn_tensor_load_to_lds(g0, g1, gz, gz, gz8, 0);
#else
  __builtin_amdgcn_tensor_load_to_lds(g0, g1, gz, gz, 0);
#endif
}

// ---------------------------------------------------------------------------
// Kernel 1: fused max-pool + depthwise 5x5 correlation + diff -> NHWC bf16.
// ---------------------------------------------------------------------------
__global__ __launch_bounds__(256) void k_dwcorr(const float* __restrict__ tf,
                                                const float* __restrict__ img,
                                                __bf16* __restrict__ corrA,
                                                __bf16* __restrict__ diffA) {
  __shared__ float plane[HWN];
  __shared__ float kern[25];
  const int tid = threadIdx.x;
  const int blk = blockIdx.x;
  const int b  = blk >> 10;
  const int tc = blk & 1023;           // t*512 + c
  const int c  = tc & 511;
  const float* ip = img + ((size_t)b * CC + c) * HWN;
  for (int i = tid; i < HWN; i += 256) plane[i] = ip[i];
  if (tid < 25) kern[tid] = tf[((size_t)(b * TT) + (tc >> 9)) * CC * 25 + (size_t)c * 25 + tid];
  __syncthreads();

  float pooled = kern[0];
  #pragma unroll
  for (int j = 1; j < 25; ++j) pooled = fmaxf(pooled, kern[j]);

  const size_t ob = (size_t)b * HWN * CIN + tc;   // NHWC scatter base
  for (int p = tid; p < HWN; p += 256) {
    const int h = p / WW, w = p % WW;
    float s = 0.f;
    #pragma unroll
    for (int kh = 0; kh < 5; ++kh) {
      const int y = h + kh - 2;
      if (y < 0 || y >= HH) continue;
      #pragma unroll
      for (int kw = 0; kw < 5; ++kw) {
        const int x = w + kw - 2;
        if (x < 0 || x >= WW) continue;
        s += plane[y * WW + x] * kern[kh * 5 + kw];
      }
    }
    corrA[ob + (size_t)p * CIN] = (__bf16)s;
    diffA[ob + (size_t)p * CIN] = (__bf16)(plane[p] - pooled);
  }
}

// ---------------------------------------------------------------------------
// Kernel 2: repack conv weights fp32 OIHW -> bf16 [tap][cout][cin]
// ---------------------------------------------------------------------------
__global__ __launch_bounds__(256) void k_packw(const float* __restrict__ w,
                                               __bf16* __restrict__ wp) {
  const int idx = blockIdx.x * 256 + threadIdx.x;
  const int total = 9 * COUT * CIN;
  if (idx >= total) return;
  const int r   = idx / (COUT * CIN);
  const int rem = idx - r * (COUT * CIN);
  const int n   = rem / CIN;
  const int ci  = rem - n * CIN;
  wp[idx] = (__bf16)w[((size_t)n * CIN + ci) * 9 + r];
}

// ---------------------------------------------------------------------------
// Kernel 3: 3x3 conv (pad 1) as tap-major implicit GEMM, NHWC, double-buffered
// pipeline. B tile (128x32) fetched by the Tensor Data Mover (TENSORcnt, wave 0
// issues, HW pads LDS rows to 80B); A tile (64x32) staged with b128 loads +
// branchless cndmask padding. One barrier per K-step.
// ---------------------------------------------------------------------------
#define ASTR 40
#define BSTR 40

template <int OCT, int OCOFF>
__global__ __launch_bounds__(256) void k_conv3x3(const __bf16* __restrict__ Act,
                                                 const __bf16* __restrict__ Wp,
                                                 const float* __restrict__ bias,
                                                 __bf16* __restrict__ Out) {
  __shared__ __bf16 Als[2][64 * ASTR];
  __shared__ __bf16 Bls[2][128 * BSTR];

  const int tid    = threadIdx.x;
  const int lane   = tid & 31;
  const int laneN  = lane & 15;
  const int laneHi = lane >> 4;
  const int wid    = tid >> 5;
  const int waveM  = wid >> 2;             // 0..1
  const int waveN  = wid & 3;              // 0..3
  const bool isW0  = (wid == 0);           // wave-uniform
  const int mb0 = blockIdx.x * 64;
  const int nb0 = blockIdx.y * 128;
  const int b   = blockIdx.z;

  const __bf16* Ab = Act + (size_t)b * HWN * CIN;
  const unsigned blsBase[2] = { (unsigned)(uintptr_t)&Bls[0][0],
                                (unsigned)(uintptr_t)&Bls[1][0] };

  // per-thread A staging geometry: exactly one 16B chunk of the 64x32 tile
  const int miA = tid >> 2;                // row 0..63
  const int seg = (tid & 3) * 8;           // cin sub-offset (elements)

  v8f acc[2][2];
  #pragma unroll
  for (int i = 0; i < 2; ++i)
    #pragma unroll
    for (int j = 0; j < 2; ++j) { v8f z{}; acc[i][j] = z; }

  for (int r = 0; r < 9; ++r) {
    const int kh = r / 3, kw = r % 3;

    // hoisted per-tap geometry
    const __bf16* WpR = Wp + (size_t)(r * COUT + nb0) * CIN;
    const int m = mb0 + miA;
    const int hh = m / WW, ww = m - hh * WW;
    const int y = hh + kh - 1, x = ww + kw - 1;
    const bool vA = (m < HWN) & (y >= 0) & (y < HH) & (x >= 0) & (x < WW);
    const __bf16* arow = Ab + (size_t)(vA ? (y * WW + x) : 0) * CIN + seg;

    auto loadA = [&](int ks) -> v8bf {
      v8bf d = *(const v8bf*)(arow + ks * 32);
      v8bf z{};
      return vA ? d : z;
    };
    auto storeA = [&](v8bf d, int buf) {
      *(v8bf*)&Als[buf][miA * ASTR + seg] = d;
    };

    // ---- prologue: stage K-step 0 into buffer 0 ----
    {
      if (isW0) tdm_load_B(WpR, blsBase[0]);
      storeA(loadA(0), 0);
      if (isW0) __builtin_amdgcn_s_wait_tensorcnt((short)0);
      __syncthreads();
    }

    #pragma unroll 2
    for (int ks = 0; ks < NKS; ++ks) {
      const int cur = ks & 1;
      const int nxt = cur ^ 1;
      const bool have_next = (ks + 1 < NKS);

      v8bf an{};
      if (have_next) {           // stage next step while this one computes
        if (isW0) tdm_load_B(WpR + (ks + 1) * 32, blsBase[nxt]);
        an = loadA(ks + 1);
      }

      // fragments of current step
      v16bf afrag[2];
      #pragma unroll
      for (int tm = 0; tm < 2; ++tm) {
        const __bf16* row = &Als[cur][(waveM * 32 + tm * 16 + laneN) * ASTR];
        afrag[tm] = cat8(*(const v8bf*)(row + laneHi * 8),
                         *(const v8bf*)(row + 16 + laneHi * 8));
      }
      v16bf bfrag[2];
      #pragma unroll
      for (int tn = 0; tn < 2; ++tn) {
        const __bf16* row = &Bls[cur][(waveN * 32 + tn * 16 + laneN) * BSTR];
        bfrag[tn] = cat8(*(const v8bf*)(row + laneHi * 8),
                         *(const v8bf*)(row + 16 + laneHi * 8));
      }

      #pragma unroll
      for (int tm = 0; tm < 2; ++tm)
        #pragma unroll
        for (int tn = 0; tn < 2; ++tn)
          acc[tm][tn] = __builtin_amdgcn_wmma_f32_16x16x32_bf16(
              false, afrag[tm], false, bfrag[tn],
              (short)0, acc[tm][tn], false, false);

      if (have_next) {
        storeA(an, nxt);
        if (isW0) __builtin_amdgcn_s_wait_tensorcnt((short)0);
        __syncthreads();
      }
    }
  }

  // ---- epilogue: bias + ReLU, NHWC bf16, constant-stride addressing ----
  const bool interior = (mb0 + 64 <= HWN);   // uniform branch
  #pragma unroll
  for (int tm = 0; tm < 2; ++tm) {
    #pragma unroll
    for (int tn = 0; tn < 2; ++tn) {
      const int n = nb0 + waveN * 32 + tn * 16 + laneN;
      const int m0 = mb0 + waveM * 32 + tm * 16 + laneHi * 8;
      const float bs = bias[n];
      __bf16* op = Out + ((size_t)b * HWN + m0) * OCT + OCOFF + n;
      v8f a = acc[tm][tn];
      if (interior) {
        #pragma unroll
        for (int v = 0; v < 8; ++v)
          op[(size_t)v * OCT] = (__bf16)fmaxf(a[v] + bs, 0.f);
      } else {
        #pragma unroll
        for (int v = 0; v < 8; ++v)
          if (m0 + v < HWN) op[(size_t)v * OCT] = (__bf16)fmaxf(a[v] + bs, 0.f);
      }
    }
  }
}

// ---------------------------------------------------------------------------
// Kernel 4: 1x1 heads + paired softmax over NHWC emb.
// emb tile staged LDS-direct with global_load_async_to_lds_b128 (ASYNCcnt);
// tail rows are clamped duplicates and never read.
// ---------------------------------------------------------------------------
#define ESTR 520
__global__ __launch_bounds__(256) void k_head(const __bf16* __restrict__ emb,
                                              const float* __restrict__ sw,
                                              const float* __restrict__ sb,
                                              const float* __restrict__ bw,
                                              const float* __restrict__ bbias,
                                              float* __restrict__ out) {
  __shared__ __bf16 tile[64 * ESTR];
  const int tid = threadIdx.x;
  const int pos0 = blockIdx.x * 64;
  const int b = blockIdx.y;
  const __bf16* ep = emb + (size_t)b * HWN * COUT;
  const unsigned t0 = (unsigned)(uintptr_t)tile;

  #pragma unroll
  for (int i = 0; i < 16; ++i) {               // 4096 async 16B chunks
    const int q = tid + i * 256;
    const int p = q >> 6;
    const int c0 = (q & 63) * 8;
    const int pos = pos0 + p;
    const int pc = (pos < HWN) ? pos : (HWN - 1);
    const __bf16* gp = ep + (size_t)pc * COUT + c0;
    const unsigned lds = t0 + (unsigned)(p * ESTR + c0) * 2u;
    asm volatile("global_load_async_to_lds_b128 %0, %1, off"
                 :: "v"(lds), "v"(gp) : "memory");
  }
  asm volatile("s_wait_asynccnt 0x0" ::: "memory");
  __syncthreads();

  const int p = tid & 63;
  const int grp = tid >> 6;          // 0..3, uniform per wave
  const int pos = pos0 + p;
  if (pos >= HWN) return;
  const __bf16* row = tile + p * ESTR;

  for (int oc = grp; oc < 54; oc += 4) {
    if (oc < 18) {
      const int pch = (oc < 9) ? oc + 9 : oc - 9;
      float s = sb[oc], s2 = sb[pch];
      const float* w0 = sw + (size_t)oc * COUT;
      const float* w1 = sw + (size_t)pch * COUT;
      #pragma unroll 8
      for (int c = 0; c < COUT; ++c) {
        const float e = (float)row[c];
        s  += e * w0[c];
        s2 += e * w1[c];
      }
      out[((size_t)b * 18 + oc) * HWN + pos] = 1.f / (1.f + __expf(s2 - s));
    } else {
      const int bo = oc - 18;
      float s = bbias[bo];
      const float* w0 = bw + (size_t)bo * COUT;
      #pragma unroll 8
      for (int c = 0; c < COUT; ++c) s += (float)row[c] * w0[c];
      out[CLS_TOTAL + ((size_t)b * 36 + bo) * HWN + pos] = s;
    }
  }
}

// ---------------------------------------------------------------------------
extern "C" void kernel_launch(void* const* d_in, const int* in_sizes, int n_in,
                              void* d_out, int out_size, void* d_ws, size_t ws_size,
                              hipStream_t stream) {
  const float* tf      = (const float*)d_in[0];
  const float* img     = (const float*)d_in[1];
  const float* corr_w  = (const float*)d_in[2];
  const float* corr_b  = (const float*)d_in[3];
  const float* diff_w  = (const float*)d_in[4];
  const float* diff_b  = (const float*)d_in[5];
  const float* emb_w   = (const float*)d_in[6];
  const float* emb_b   = (const float*)d_in[7];
  const float* score_w = (const float*)d_in[8];
  const float* score_b = (const float*)d_in[9];
  const float* bbox_w  = (const float*)d_in[10];
  const float* bbox_b  = (const float*)d_in[11];
  float* out = (float*)d_out;

  char* ws = (char*)d_ws;
  size_t off = 0;
  auto alloc = [&](size_t bytes) -> void* {
    void* p = ws + off;
    off += (bytes + 255) & ~(size_t)255;
    return p;
  };
  const size_t actB = (size_t)BB * HWN * CIN * sizeof(__bf16);   // 31.1 MB NHWC
  const size_t wpB  = (size_t)9 * COUT * CIN * sizeof(__bf16);   //  9.4 MB
  __bf16* corrA  = (__bf16*)alloc(actB);
  __bf16* diffA  = (__bf16*)alloc(actB);
  __bf16* concat = (__bf16*)alloc(actB);
  __bf16* embB   = (__bf16*)alloc((size_t)BB * HWN * COUT * sizeof(__bf16));
  __bf16* wpC    = (__bf16*)alloc(wpB);
  __bf16* wpD    = (__bf16*)alloc(wpB);
  __bf16* wpE    = (__bf16*)alloc(wpB);

  k_dwcorr<<<BB * CIN, 256, 0, stream>>>(tf, img, corrA, diffA);

  const int wtot = 9 * COUT * CIN;
  k_packw<<<(wtot + 255) / 256, 256, 0, stream>>>(corr_w, wpC);
  k_packw<<<(wtot + 255) / 256, 256, 0, stream>>>(diff_w, wpD);
  k_packw<<<(wtot + 255) / 256, 256, 0, stream>>>(emb_w, wpE);

  dim3 cgrid((HWN + 63) / 64, COUT / 128, BB);   // 30 x 4 x 8
  k_conv3x3<CIN, 0>  <<<cgrid, 256, 0, stream>>>(corrA,  wpC, corr_b, concat);
  k_conv3x3<CIN, 512><<<cgrid, 256, 0, stream>>>(diffA,  wpD, diff_b, concat);
  k_conv3x3<COUT, 0> <<<cgrid, 256, 0, stream>>>(concat, wpE, emb_b,  embB);

  k_head<<<dim3((HWN + 63) / 64, BB), 256, 0, stream>>>(embB, score_w, score_b,
                                                        bbox_w, bbox_b, out);
}
